// VQVAEEncoder_61237643706885
// MI455X (gfx1250) — matrix operand order repaired and checked by reading, not compiled, not run on hardware
//
#include <hip/hip_runtime.h>

// ---------------------------------------------------------------------------
// VQ-VAE transformer encoder for MI455X (gfx1250, wave32, WMMA).
// All GEMM-class math uses v_wmma_f32_16x16x32_f16 (f16 in, f32 accum).
// BOTH operand classes are kept in WMMA fragment-major layout:
//   - weights/codebook: B-fragment-major (pre-shuffled once per call)
//   - f16 activations (h16,q,k,v,o,mid): A-fragment-major (written shuffled)
// so every WMMA operand load is one contiguous 32B lane slot = 2x b128 loads.
// ---------------------------------------------------------------------------

typedef _Float16 f16;
typedef __attribute__((ext_vector_type(16))) _Float16 v16h;
typedef __attribute__((ext_vector_type(8)))  float    v8f;

#define B_   8
#define T_   2048
#define DIN_ 12
#define D_   256
#define K_   1024
#define H_   4
#define DK_  64
#define L_   4
#define M_   (B_ * T_)        // 16384 tokens
#define KC_  (T_ / 32)        // 64 key chunks per sequence

__device__ __forceinline__ v8f wmma_f16(v16h a, v16h b, v8f c) {
  return __builtin_amdgcn_wmma_f32_16x16x32_f16(false, a, false, b, (short)0, c,
                                                false, false);
}

// Fragment load (A or B): 512 halves per 16x32 (A) / 32x16 (B) tile,
// each lane's 16 halves contiguous (32B aligned) -> 2x b128 loads.
__device__ __forceinline__ v16h load_frag(const f16* F, size_t tile) {
  int lane = (int)(threadIdx.x & 31u);
  return *(const v16h*)(F + tile * 512 + (size_t)lane * 16);
}

// A-fragment-major index for element (row m, col c) of an (M x 256) matrix.
// Per ISA A layout: lane = (m&15) | hf<<4 with hf=(c>>3)&1 (K 8-15/24-31 in
// upper lanes); vector elem j = ((c>>4)&1)*8 + (c&7); tile = (m/16)*8 + c/32.
__device__ __forceinline__ size_t afrag_idx(int m, int c) {
  int l = (m & 15) | (((c >> 3) & 1) << 4);
  int j = (((c >> 4) & 1) << 3) | (c & 7);
  return (((size_t)((m >> 4) * 8 + (c >> 5)) * 32 + l) << 4) | j;
}

// ---------------------------------------------------------------------------
// Shuffle kernels
// ---------------------------------------------------------------------------
// Weights / codebook -> B-fragment-major:
// dst[((kt*NT+nt)*32 + lane)*16 + j] holds
//   trans==0: src[(kt*32 + (j>>3)*16 + (lane>>4)*8 + (j&7))*ld + nt*16 + (lane&15)]
//   trans==1: src[(nt*16 + (lane&15))*ld + kt*32 + (j>>3)*16 + (lane>>4)*8 + (j&7)]
__global__ void wshuf_kernel(const float* __restrict__ src, f16* __restrict__ dst,
                             int NT, int ld, int total, int trans) {
  int idx = blockIdx.x * 256 + threadIdx.x;
  if (idx >= total) return;
  int j = idx & 15, l = (idx >> 4) & 31, t = idx >> 9;
  int nt = t % NT, kt = t / NT;
  int k = kt * 32 + ((j >> 3) << 4) + ((l >> 4) << 3) + (j & 7);
  int n = nt * 16 + (l & 15);
  float v = trans ? src[(size_t)n * ld + k] : src[(size_t)k * ld + n];
  dst[idx] = (f16)v;
}

// K fragments for S = Q K^T: per (b,h,kc) 4 frags f = dkhalf*2 + keyhalf;
// element (lane,j): key = kc*32 + keyhalf*16 + (l&15),
//                   dk  = dkhalf*32 + (j>>3)*16 + (l>>4)*8 + (j&7).   (kb is A-frag-major)
__global__ void kshuf_kernel(const f16* __restrict__ kb, f16* __restrict__ kf) {
  int idx = blockIdx.x * 256 + threadIdx.x;             // M_*D_ threads
  int j  = idx & 15;
  int l  = (idx >> 4) & 31;
  int f  = (idx >> 9) & 3;
  int kc = (idx >> 11) & (KC_ - 1);
  int h  = (idx >> 17) & (H_ - 1);
  int b  = idx >> 19;
  int key = kc * 32 + ((f & 1) << 4) + (l & 15);
  int dk  = ((f >> 1) << 5) + ((j >> 3) << 4) + ((l >> 4) << 3) + (j & 7);
  kf[idx] = kb[afrag_idx(b * T_ + key, h * DK_ + dk)];
}

// V fragments for P V: per (b,h,kc) 4 frags f = dk-tile;
// element (lane,j): key = kc*32 + (j>>3)*16 + (l>>4)*8 + (j&7), dk = f*16 + (l&15).
__global__ void vshuf_kernel(const f16* __restrict__ vb, f16* __restrict__ vf) {
  int idx = blockIdx.x * 256 + threadIdx.x;             // M_*D_ threads
  int j  = idx & 15;
  int l  = (idx >> 4) & 31;
  int f  = (idx >> 9) & 3;
  int kc = (idx >> 11) & (KC_ - 1);
  int h  = (idx >> 17) & (H_ - 1);
  int b  = idx >> 19;
  int key = kc * 32 + ((j >> 3) << 4) + ((l >> 4) << 3) + (j & 7);
  int dk  = (f << 4) + (l & 15);
  vf[idx] = vb[afrag_idx(b * T_ + key, h * DK_ + dk)];
}

// ---------------------------------------------------------------------------
// Elementwise helpers
// ---------------------------------------------------------------------------
__global__ void zerof_kernel(float* __restrict__ p, int n) {
  int i = blockIdx.x * 256 + threadIdx.x;
  if (i < n) p[i] = 0.f;
}

// h = x @ W_in + b_in + sinusoid(t, D); writes f32 row-major + f16 A-frag
__global__ void in_proj_kernel(const float* __restrict__ x, const float* __restrict__ Wi,
                               const float* __restrict__ bi, float* __restrict__ Hb,
                               f16* __restrict__ H16) {
  int idx = blockIdx.x * 256 + threadIdx.x;             // exactly M_*D_ threads
  int row = idx >> 8, d = idx & 255;
  int t = row & (T_ - 1);
  const float* xr = x + row * DIN_;
  float s = bi[d];
#pragma unroll
  for (int i = 0; i < DIN_; ++i) s += xr[i] * Wi[i * D_ + d];
  float ang = (float)t * __expf(-(float)(d >> 1) * 0.07195578415f); // ln(1e4)/128
  s += (d & 1) ? __cosf(ang) : __sinf(ang);
  Hb[idx] = s;
  H16[afrag_idx(row, d)] = (f16)s;
}

// ---------------------------------------------------------------------------
// GEMM: C(f16, A-frag-major) = A(f16 A-frag-major) @ Wf(B-frag-major) + bias.
// One wave per 16x16 tile. kts = Kdim/32 tiles along K.
// ---------------------------------------------------------------------------
__global__ void gemm_kernel(const f16* __restrict__ A, const f16* __restrict__ Wf,
                            const float* __restrict__ bias, f16* __restrict__ C,
                            int ntn, int kts, int relu) {
  int m0 = (blockIdx.x / ntn) << 4;
  int nt = blockIdx.x % ntn;
  int lane = (int)(threadIdx.x & 31u);
  int n = lane & 15, hf = lane >> 4;
  v8f acc = {};
  for (int kt = 0; kt < kts; ++kt)
    acc = wmma_f16(load_frag(A, (size_t)(m0 >> 4) * kts + kt),
                   load_frag(Wf, (size_t)kt * ntn + nt), acc);
  float bv = bias[nt * 16 + n];
  int c = nt * 16 + n;
#pragma unroll
  for (int r = 0; r < 8; ++r) {
    int m = m0 + (hf << 3) + r;
    float v = acc[r] + bv;
    if (relu) v = fmaxf(v, 0.f);
    C[afrag_idx(m, c)] = (f16)v;
  }
}

// ---------------------------------------------------------------------------
// Flash attention: one wave per (q-tile, head, batch). Streams keys in chunks
// of 32 with online softmax; Q/K/V all fragment-major (b128 loads); P staged
// through LDS in A-fragment order (b128 ds loads); next chunk prefetched.
// ---------------------------------------------------------------------------
__global__ void attn_kernel(const f16* __restrict__ Q, const f16* __restrict__ Kf,
                            const f16* __restrict__ Vf, f16* __restrict__ O) {
  __shared__ f16 pbuf[512];                             // one 16x32 A-frag tile
  int qt = blockIdx.x, hh = blockIdx.y, b = blockIdx.z;
  int lane = (int)(threadIdx.x & 31u);
  int n = lane & 15, hf = lane >> 4;
  v16h qa0 = load_frag(Q, (size_t)(b * (T_ >> 4) + qt) * 8 + hh * 2);
  v16h qa1 = load_frag(Q, (size_t)(b * (T_ >> 4) + qt) * 8 + hh * 2 + 1);
  float mrow[8], lrow[8];
  v8f oacc[4] = {};
#pragma unroll
  for (int r = 0; r < 8; ++r) { mrow[r] = -1e30f; lrow[r] = 0.f; }
  const float scale = 0.125f;                           // 1/sqrt(64)
  size_t bh = (size_t)(b * H_ + hh) * KC_;
  // P value for (row m, key kk) goes to A-frag slot:
  //   lane = m | ((kk>>3 & 1)<<4), elem = ((kk>>4)&1)*8 + (kk&7)
  int pl0 = ((n >> 3) & 1) << 4;                        // lane half from key n
  for (int kc = 0; kc < KC_; ++kc) {
    const f16* kfb = Kf + (bh + kc) * 4 * 512;
    const f16* vfb = Vf + (bh + kc) * 4 * 512;
    if (kc + 1 < KC_) {                                 // global_prefetch_b8
      __builtin_prefetch(kfb + 4 * 512 + lane * 16, 0, 0);
      __builtin_prefetch(vfb + 4 * 512 + lane * 16, 0, 0);
    }
    v8f s0 = {}, s1 = {};
    s0 = wmma_f16(qa0, load_frag(kfb, 0), s0);          // dk 0-31 , keys 0-15
    s0 = wmma_f16(qa1, load_frag(kfb, 2), s0);          // dk 32-63, keys 0-15
    s1 = wmma_f16(qa0, load_frag(kfb, 1), s1);          // dk 0-31 , keys 16-31
    s1 = wmma_f16(qa1, load_frag(kfb, 3), s1);          // dk 32-63, keys 16-31
    float esc[8];
#pragma unroll
    for (int r = 0; r < 8; ++r) {
      float v0 = s0[r] * scale, v1 = s1[r] * scale;
      float mx = fmaxf(v0, v1);
#pragma unroll
      for (int off = 1; off < 16; off <<= 1) mx = fmaxf(mx, __shfl_xor(mx, off));
      float mnew = fmaxf(mrow[r], mx);
      esc[r] = __expf(mrow[r] - mnew);
      mrow[r] = mnew;
      float p0 = __expf(v0 - mnew);
      float p1 = __expf(v1 - mnew);
      float ps = p0 + p1;
#pragma unroll
      for (int off = 1; off < 16; off <<= 1) ps += __shfl_xor(ps, off);
      lrow[r] = lrow[r] * esc[r] + ps;
      int m = (hf << 3) + r;
      int slot = ((m | pl0) << 4) | (n & 7);
      pbuf[slot]     = (f16)p0;                         // key n   (elem j = n&7)
      pbuf[slot + 8] = (f16)p1;                         // key n+16 (elem j+8)
    }
    __syncthreads();
#pragma unroll
    for (int t = 0; t < 4; ++t)
#pragma unroll
      for (int r = 0; r < 8; ++r) oacc[t][r] *= esc[r];
    v16h pa = *(const v16h*)(pbuf + lane * 16);         // 2x ds b128
#pragma unroll
    for (int t = 0; t < 4; ++t)
      oacc[t] = wmma_f16(pa, load_frag(vfb, t), oacc[t]);
    __syncthreads();                                    // WAR on pbuf
  }
#pragma unroll
  for (int t = 0; t < 4; ++t)
#pragma unroll
    for (int r = 0; r < 8; ++r) {
      int m = (hf << 3) + r;
      float v = oacc[t][r] / lrow[r];
      O[afrag_idx(b * T_ + qt * 16 + m, hh * DK_ + t * 16 + n)] = (f16)v;
    }
}

// ---------------------------------------------------------------------------
// Fused: H = LayerNorm(H + A(A-frag) @ Wf(B-frag 256x256) + bias; g, b)
// 4 waves per block; wave w owns 64 output columns; LN via LDS tile.
// Writes f32 row-major H and f16 A-frag shadow H16.
// ---------------------------------------------------------------------------
__global__ void proj_ln_kernel(const f16* __restrict__ A, const f16* __restrict__ Wf,
                               const float* __restrict__ bias, const float* __restrict__ g,
                               const float* __restrict__ bln, float* __restrict__ Hb,
                               f16* __restrict__ H16) {
  __shared__ float tile[16][D_ + 1];
  int m0 = blockIdx.x << 4;
  int wave = (int)(threadIdx.x >> 5u);
  int lane = (int)(threadIdx.x & 31u);
  int n = lane & 15, hf = lane >> 4;
  v8f acc[4] = {};
  for (int kt = 0; kt < 8; ++kt) {
    v16h a = load_frag(A, (size_t)(m0 >> 4) * 8 + kt);
#pragma unroll
    for (int t = 0; t < 4; ++t)
      acc[t] = wmma_f16(a, load_frag(Wf, kt * 16 + wave * 4 + t), acc[t]);
  }
#pragma unroll
  for (int t = 0; t < 4; ++t) {
    int c = (wave * 4 + t) * 16 + n;
#pragma unroll
    for (int r = 0; r < 8; ++r) {
      int m = (hf << 3) + r;
      tile[m][c] = acc[t][r] + bias[c] + Hb[(size_t)(m0 + m) * D_ + c];
    }
  }
  __syncthreads();
  int row = (int)(threadIdx.x >> 3u);                   // 0..15
  int seg = (int)(threadIdx.x & 7u);                    // 8 segs x 32 cols
  float s = 0.f, s2 = 0.f;
#pragma unroll
  for (int i = 0; i < 32; ++i) {
    float v = tile[row][seg * 32 + i];
    s += v; s2 += v * v;
  }
#pragma unroll
  for (int off = 1; off < 8; off <<= 1) { s += __shfl_xor(s, off); s2 += __shfl_xor(s2, off); }
  float mean = s * (1.f / 256.f);
  float var  = s2 * (1.f / 256.f) - mean * mean;
  float rstd = rsqrtf(var + 1e-5f);
#pragma unroll
  for (int i = 0; i < 32; ++i) {
    int c = seg * 32 + i;
    float v = (tile[row][c] - mean) * rstd * g[c] + bln[c];
    Hb[(size_t)(m0 + row) * D_ + c] = v;
    H16[afrag_idx(m0 + row, c)] = (f16)v;
  }
}

// ---------------------------------------------------------------------------
// VQ stage
// ---------------------------------------------------------------------------
__global__ void cnorm_kernel(const float* __restrict__ cb, float* __restrict__ cn) {
  int k = blockIdx.x;
  float s = 0.f;
  for (int d = (int)threadIdx.x; d < D_; d += 32) {
    float v = cb[(size_t)k * D_ + d];
    s += v * v;
  }
#pragma unroll
  for (int off = 1; off < 32; off <<= 1) s += __shfl_xor(s, off);
  if (threadIdx.x == 0) cn[k] = s;
}

// argmin_k ( ||c_k||^2 - 2 z·c_k ) — monotone in the true distance.
// Z in A-frag-major; codebook B-frag-major (transposed), NT = K_/16 = 64.
__global__ void vq_argmin_kernel(const f16* __restrict__ Z16, const f16* __restrict__ CBf,
                                 const float* __restrict__ cnorm, int* __restrict__ ids) {
  int m0 = blockIdx.x << 4;
  int lane = (int)(threadIdx.x & 31u);
  int n = lane & 15, hf = lane >> 4;
  v16h afr[8];
#pragma unroll
  for (int ks = 0; ks < 8; ++ks)
    afr[ks] = load_frag(Z16, (size_t)(m0 >> 4) * 8 + ks);
  float best[8]; int bidx[8];
#pragma unroll
  for (int r = 0; r < 8; ++r) { best[r] = 1e30f; bidx[r] = 0; }
  for (int nt = 0; nt < (K_ >> 4); ++nt) {
    v8f s = {};
#pragma unroll
    for (int ks = 0; ks < 8; ++ks)
      s = wmma_f16(afr[ks], load_frag(CBf, (size_t)ks * (K_ >> 4) + nt), s);
    float cn = cnorm[nt * 16 + n];
#pragma unroll
    for (int r = 0; r < 8; ++r) {
      float d = cn - 2.f * s[r];
      if (d < best[r]) { best[r] = d; bidx[r] = nt * 16 + n; }
    }
  }
#pragma unroll
  for (int off = 1; off < 16; off <<= 1) {
#pragma unroll
    for (int r = 0; r < 8; ++r) {
      float ov = __shfl_xor(best[r], off);
      int oi = __shfl_xor(bidx[r], off);
      if (ov < best[r] || (ov == best[r] && oi < bidx[r])) { best[r] = ov; bidx[r] = oi; }
    }
  }
#pragma unroll
  for (int r = 0; r < 8; ++r)
    if (n == r) ids[m0 + (hf << 3) + r] = bidx[r];
}

__global__ void vq_scatter_kernel(const float* __restrict__ Z, const int* __restrict__ ids,
                                  float* __restrict__ sums, float* __restrict__ counts) {
  int idx = blockIdx.x * 256 + threadIdx.x;             // M_*D_ threads
  int row = idx >> 8, d = idx & 255;
  int id = ids[row];
  atomicAdd(&sums[(size_t)id * D_ + d], Z[idx]);
  if (d == 0) atomicAdd(&counts[id], 1.0f);
}

__global__ void vq_gather_kernel(const float* __restrict__ Z, const float* __restrict__ cb,
                                 const int* __restrict__ ids, float* __restrict__ oze,
                                 float* __restrict__ ozq) {
  int idx = blockIdx.x * 256 + threadIdx.x;             // M_*D_ threads
  int row = idx >> 8, d = idx & 255;
  oze[idx] = Z[idx];
  ozq[idx] = cb[(size_t)ids[row] * D_ + d];
}

__global__ void vq_smooth_kernel(const float* __restrict__ ecs, const float* __restrict__ counts,
                                 float* __restrict__ smoothed) {
  __shared__ float sh[32];
  int k = (int)threadIdx.x;                             // K_ threads
  float ncs = ecs[k] * 0.99f + 0.01f * counts[k];
  float s = ncs;
#pragma unroll
  for (int off = 16; off; off >>= 1) s += __shfl_xor(s, off);
  if ((k & 31) == 0) sh[k >> 5] = s;
  __syncthreads();
  if (k < 32) {
    float v = sh[k];
#pragma unroll
    for (int off = 16; off; off >>= 1) v += __shfl_xor(v, off);
    if (k == 0) sh[0] = v;
  }
  __syncthreads();
  float n_tot = sh[0];
  smoothed[k] = (ncs + 1e-5f) / (n_tot + (float)K_ * 1e-5f) * n_tot;
}

__global__ void vq_update_kernel(const float* __restrict__ emaw, const float* __restrict__ sums,
                                 const float* __restrict__ smoothed, float* __restrict__ ocb) {
  int idx = blockIdx.x * 256 + threadIdx.x;             // K_*D_ threads
  int k = idx >> 8;
  float nw = emaw[idx] * 0.99f + 0.01f * sums[idx];
  ocb[idx] = nw / smoothed[k];
}

// ---------------------------------------------------------------------------
// Launch
// ---------------------------------------------------------------------------
extern "C" void kernel_launch(void* const* d_in, const int* in_sizes, int n_in,
                              void* d_out, int out_size, void* d_ws, size_t ws_size,
                              hipStream_t stream) {
  (void)in_sizes; (void)n_in; (void)out_size; (void)ws_size;
  const float* x    = (const float*)d_in[0];
  const float* W_in = (const float*)d_in[1];
  const float* b_in = (const float*)d_in[2];
  const float* Wq   = (const float*)d_in[3];
  const float* Wk   = (const float*)d_in[4];
  const float* Wv   = (const float*)d_in[5];
  const float* Wo   = (const float*)d_in[6];
  const float* bq   = (const float*)d_in[7];
  const float* bk   = (const float*)d_in[8];
  const float* bv   = (const float*)d_in[9];
  const float* bo   = (const float*)d_in[10];
  const float* ln1g = (const float*)d_in[11];
  const float* ln1b = (const float*)d_in[12];
  const float* ln2g = (const float*)d_in[13];
  const float* ln2b = (const float*)d_in[14];
  const float* W1   = (const float*)d_in[15];
  const float* b1   = (const float*)d_in[16];
  const float* W2   = (const float*)d_in[17];
  const float* b2   = (const float*)d_in[18];
  const float* cb   = (const float*)d_in[19];
  const float* emaw = (const float*)d_in[20];
  const float* emacs= (const float*)d_in[21];

  // workspace carve (all 256B aligned)
  char* w = (char*)d_ws;
  auto carve = [&](size_t bytes) { char* p = w; w += (bytes + 255) & ~(size_t)255; return p; };
  float* h    = (float*)carve((size_t)M_ * D_ * 4);
  f16*   h16  = (f16*)carve((size_t)M_ * D_ * 2);   // A-frag-major
  f16*   qb   = (f16*)carve((size_t)M_ * D_ * 2);   // A-frag-major
  f16*   kb   = (f16*)carve((size_t)M_ * D_ * 2);   // A-frag-major
  f16*   vb   = (f16*)carve((size_t)M_ * D_ * 2);   // A-frag-major
  f16*   kfr  = (f16*)carve((size_t)M_ * D_ * 2);   // K B-frag-major
  f16*   vfr  = (f16*)carve((size_t)M_ * D_ * 2);   // V B-frag-major
  f16*   ob   = (f16*)carve((size_t)M_ * D_ * 2);   // A-frag-major
  f16*   midb = (f16*)carve((size_t)M_ * D_ * 2);   // A-frag-major
  const int WSZ = D_ * D_;                // 65536 per layer
  f16* wq16 = (f16*)carve((size_t)L_ * WSZ * 2);
  f16* wk16 = (f16*)carve((size_t)L_ * WSZ * 2);
  f16* wv16 = (f16*)carve((size_t)L_ * WSZ * 2);
  f16* wo16 = (f16*)carve((size_t)L_ * WSZ * 2);
  f16* w116 = (f16*)carve((size_t)L_ * WSZ * 2);
  f16* w216 = (f16*)carve((size_t)L_ * WSZ * 2);
  f16* cbf  = (f16*)carve((size_t)K_ * D_ * 2);
  float* cnorm = (float*)carve(K_ * 4);
  int*   ids   = (int*)carve(M_ * 4);
  float* sums  = (float*)carve(((size_t)K_ * D_ + K_) * 4);
  float* counts = sums + (size_t)K_ * D_;
  float* smoothed = (float*)carve(K_ * 4);

  // 1) pre-shuffle weights + codebook into B-fragment-major f16
  const int WBLK = WSZ / 256;             // 256 blocks per layer-tensor
  for (int l = 0; l < L_; ++l) {
    wshuf_kernel<<<WBLK, 256, 0, stream>>>(Wq + (size_t)l * WSZ, wq16 + (size_t)l * WSZ, 16, D_, WSZ, 0);
    wshuf_kernel<<<WBLK, 256, 0, stream>>>(Wk + (size_t)l * WSZ, wk16 + (size_t)l * WSZ, 16, D_, WSZ, 0);
    wshuf_kernel<<<WBLK, 256, 0, stream>>>(Wv + (size_t)l * WSZ, wv16 + (size_t)l * WSZ, 16, D_, WSZ, 0);
    wshuf_kernel<<<WBLK, 256, 0, stream>>>(Wo + (size_t)l * WSZ, wo16 + (size_t)l * WSZ, 16, D_, WSZ, 0);
    wshuf_kernel<<<WBLK, 256, 0, stream>>>(W1 + (size_t)l * WSZ, w116 + (size_t)l * WSZ, 16, D_, WSZ, 0);
    wshuf_kernel<<<WBLK, 256, 0, stream>>>(W2 + (size_t)l * WSZ, w216 + (size_t)l * WSZ, 16, D_, WSZ, 0);
  }
  wshuf_kernel<<<(K_ * D_) / 256, 256, 0, stream>>>(cb, cbf, K_ >> 4, D_, K_ * D_, 1); // transposed

  // 2) input projection + positional encoding
  in_proj_kernel<<<M_, 256, 0, stream>>>(x, W_in, b_in, h, h16);

  // 3) transformer layers
  const dim3 ggrid((M_ >> 4) * (D_ >> 4));
  const int SHBLK = (M_ * D_) / 256;
  for (int l = 0; l < L_; ++l) {
    gemm_kernel<<<ggrid, 32, 0, stream>>>(h16, wq16 + (size_t)l * WSZ, bq + l * D_, qb, 16, 8, 0);
    gemm_kernel<<<ggrid, 32, 0, stream>>>(h16, wk16 + (size_t)l * WSZ, bk + l * D_, kb, 16, 8, 0);
    gemm_kernel<<<ggrid, 32, 0, stream>>>(h16, wv16 + (size_t)l * WSZ, bv + l * D_, vb, 16, 8, 0);
    kshuf_kernel<<<SHBLK, 256, 0, stream>>>(kb, kfr);
    vshuf_kernel<<<SHBLK, 256, 0, stream>>>(vb, vfr);
    attn_kernel<<<dim3(T_ >> 4, H_, B_), 32, 0, stream>>>(qb, kfr, vfr, ob);
    proj_ln_kernel<<<M_ >> 4, 128, 0, stream>>>(ob, wo16 + (size_t)l * WSZ, bo + l * D_,
                                                ln1g + l * D_, ln1b + l * D_, h, h16);
    gemm_kernel<<<ggrid, 32, 0, stream>>>(h16, w116 + (size_t)l * WSZ, b1 + l * D_, midb, 16, 8, 1);
    proj_ln_kernel<<<M_ >> 4, 128, 0, stream>>>(midb, w216 + (size_t)l * WSZ, b2 + l * D_,
                                                ln2g + l * D_, ln2b + l * D_, h, h16);
  }

  // 4) VQ: nearest codebook + EMA update
  float* out = (float*)d_out;
  cnorm_kernel<<<K_, 32, 0, stream>>>(cb, cnorm);
  vq_argmin_kernel<<<M_ >> 4, 32, 0, stream>>>(h16, cbf, cnorm, ids);
  zerof_kernel<<<((K_ * D_ + K_) + 255) / 256, 256, 0, stream>>>(sums, K_ * D_ + K_);
  vq_scatter_kernel<<<SHBLK, 256, 0, stream>>>(h, ids, sums, counts);
  vq_gather_kernel<<<SHBLK, 256, 0, stream>>>(h, cb, ids, out, out + (size_t)M_ * D_);
  vq_smooth_kernel<<<1, K_, 0, stream>>>(emacs, counts, smoothed);
  vq_update_kernel<<<(K_ * D_) / 256, 256, 0, stream>>>(emaw, sums, smoothed,
                                                        out + (size_t)2 * M_ * D_);
}
